// SpeechSegmentSelector_41592463294645
// MI455X (gfx1250) — matrix operand order repaired and checked by reading, not compiled, not run on hardware
//
#include <hip/hip_runtime.h>

typedef _Float16 h16;
typedef __attribute__((ext_vector_type(4)))  _Float16 v4h;
typedef __attribute__((ext_vector_type(8)))  _Float16 v8h;
typedef __attribute__((ext_vector_type(16))) _Float16 v16h;
typedef __attribute__((ext_vector_type(8)))  float    v8f;

#define NB 8
#define NN 65536
#define KZF 1023
#define NH 512
#define NJ 32
#define TJLEN 2262
#define SEGX 3328
#define EPSF 1.1920929e-07f

// ---------------------------------------------------------------- kernel 1
// Per-batch max-abs normalization of X -> Xn (f32).
__global__ void k_norm(const float* __restrict__ X, float* __restrict__ Xn) {
    __shared__ float red[256];
    int b = blockIdx.x, tid = threadIdx.x;
    const float* x = X + (size_t)b * NN;
    float m = 0.f;
    for (int i = tid; i < NN; i += 256) m = fmaxf(m, fabsf(x[i]));
    red[tid] = m; __syncthreads();
    for (int s = 128; s > 0; s >>= 1) {
        if (tid < s) red[tid] = fmaxf(red[tid], red[tid + s]);
        __syncthreads();
    }
    float inv = 1.f / fmaxf(red[0], EPSF);
    for (int i = tid; i < NN; i += 256) Xn[(size_t)b * NN + i] = x[i] * inv;
}

// ---------------------------------------------------------------- kernel 2
// Per (b,h): autocovariance of segment [h*128, h*128+1023), normalized by
// max-abs. Writes f32 filters into d_out and an f16 copy (k padded to 1024,
// pad element = 0) for the WMMA conv.
__global__ void k_autocov(const float* __restrict__ Xn,
                          float* __restrict__ out, h16* __restrict__ Ff16) {
    __shared__ float s[KZF];
    __shared__ float red[256];
    int bid = blockIdx.x;
    int b = bid >> 9, h = bid & 511;
    int tid = threadIdx.x;
    const float* x = Xn + (size_t)b * NN;
    int base = h * 128;

    float psum = 0.f;
    for (int i = tid; i < KZF; i += 256) {
        int g = base + i;
        float v = (g < NN) ? x[g] : 0.f;   // zero pad past end (2*KZ pad in ref)
        s[i] = v; psum += v;
    }
    red[tid] = psum; __syncthreads();
    for (int st = 128; st > 0; st >>= 1) {
        if (tid < st) red[tid] += red[tid + st];
        __syncthreads();
    }
    float mean = red[0] * (1.f / 1023.f);
    __syncthreads();
    for (int i = tid; i < KZF; i += 256) s[i] -= mean;
    __syncthreads();

    float ac[4];
    float mloc = 0.f;
    #pragma unroll
    for (int q = 0; q < 4; ++q) {
        int l = tid + q * 256;
        float a = 0.f;
        if (l < KZF) {
            for (int t = 0; t + l < KZF; ++t) a += s[t] * s[t + l];
            a *= (1.f / 1023.f);
        }
        ac[q] = a;
        mloc = fmaxf(mloc, fabsf(a));
    }
    red[tid] = mloc; __syncthreads();
    for (int st = 128; st > 0; st >>= 1) {
        if (tid < st) red[tid] = fmaxf(red[tid], red[tid + st]);
        __syncthreads();
    }
    float inv = 1.f / fmaxf(red[0], EPSF);
    size_t ob = (size_t)(b * NH + h);
    #pragma unroll
    for (int q = 0; q < 4; ++q) {
        int l = tid + q * 256;
        if (l < KZF) {
            float v = ac[q] * inv;
            out[ob * KZF + l] = v;
            Ff16[(ob << 10) + l] = (h16)v;
        }
    }
    if (tid == 0) Ff16[(ob << 10) + 1023] = (h16)0.f;
}

// Weight of position tp (relative to e2-window start) = number of the 8
// pool-1 windows (len 512, stride 250) covering it. Pools collapse to this.
__device__ __forceinline__ float e2weight(int tp) {
    if (tp < 0 || tp >= TJLEN) return 0.f;
    int lo = (tp >= 512) ? (tp - 262) / 250 : 0;   // ceil((tp-511)/250)
    int hi = tp / 250; if (hi > 7) hi = 7;
    return (hi >= lo) ? (float)(hi - lo + 1) : 0.f;
}

// ---------------------------------------------------------------- kernel 3
// WMMA conv + fused pooling. Grid: (j=32, h-tile=32, b=8), 128 threads.
// Each workgroup: 16 filters x one e2 window (2262 t positions), full K=1023.
// D = A(F tile 16xK) x B(Toeplitz x, Kx16). Never materializes y: reduces
// w(t)*y^2 into 16 scalars -> epc[b,h,j].
__launch_bounds__(128)
__global__ void k_convpool(const float* __restrict__ Xn,
                           const h16* __restrict__ Ff16,
                           float* __restrict__ epc) {
    __shared__ __align__(16) h16 Fs[16][1024];   // 32 KB, filter tile (k=1023 is 0)
    __shared__ __align__(16) h16 Xs4[SEGX * 4];  // 26 KB, 4x duplicated x segment
    __shared__ float redE[4][16][16];            // 4 KB, deterministic reduction

    int j = blockIdx.x, hb = blockIdx.y * 16, b = blockIdx.z;
    int tid = threadIdx.x;
    int tj = 2000 * j;
    const float* x = Xn + (size_t)b * NN;

    for (int i = tid; i < 16 * 1024; i += 128) {
        int hh = i >> 10, k = i & 1023;
        Fs[hh][k] = Ff16[(((size_t)(b * NH + hb + hh)) << 10) + k];
    }
    // Xs4[4e + c] = x_seg[e + c], x_seg[i] = x[tj - 511 + i] (0 out of range)
    for (int i = tid; i < SEGX * 4; i += 128) {
        int e = i >> 2, c = i & 3;
        int g = tj - 511 + e + c;
        float v = (g >= 0 && g < NN) ? x[g] : 0.f;
        Xs4[i] = (h16)v;
    }
    __syncthreads();

    int w = tid >> 5, lane = tid & 31;
    int ln = lane & 15, lh = lane >> 4;

    float eacc[8];
    #pragma unroll
    for (int v = 0; v < 8; ++v) eacc[v] = 0.f;

    for (int tc = 0; tc < 36; ++tc) {           // 36*64 = 2304 >= 2262
        int t0 = tc * 64 + w * 16;              // per-wave t column base
        v8f acc = {};
        int bbase = t0 + ln;
        for (int kc = 0; kc < 1024; kc += 32) {
            // A fragment (16x32 f16): lane m=ln; VGPRs 0-3 = F[m, kc+16*lh..+7],
            // VGPRs 4-7 = F[m, kc+16*lh+8..+15] -> two aligned b128 LDS loads.
            v16h a;
            {
                v8h alo = *(const v8h*)&Fs[ln][kc + 16 * lh];
                v8h ahi = *(const v8h*)&Fs[ln][kc + 16 * lh + 8];
                #pragma unroll
                for (int i = 0; i < 8; ++i) { a[i] = alo[i]; a[8 + i] = ahi[i]; }
            }
            // B fragment (32x16 f16): lane n=ln, K-half lh; per-lane the 16
            // halves are x_seg[bb .. bb+15]; 4x-dup array -> 4 aligned b64 loads.
            int bb = bbase + kc + 16 * lh;
            v16h bf;
            {
                v4h b0 = *(const v4h*)&Xs4[4 * bb];
                v4h b1 = *(const v4h*)&Xs4[4 * (bb + 4)];
                v4h b2 = *(const v4h*)&Xs4[4 * (bb + 8)];
                v4h b3 = *(const v4h*)&Xs4[4 * (bb + 12)];
                #pragma unroll
                for (int i = 0; i < 4; ++i) {
                    bf[i] = b0[i]; bf[4 + i] = b1[i];
                    bf[8 + i] = b2[i]; bf[12 + i] = b3[i];
                }
            }
            acc = __builtin_amdgcn_wmma_f32_16x16x32_f16(
                false, a, false, bf, (short)0, acc, false, false);
        }
        // D layout: lane n=ln -> t = t0+ln ; VGPR v -> h_local = v + 8*lh
        float wgt = e2weight(t0 + ln);
        #pragma unroll
        for (int v = 0; v < 8; ++v) eacc[v] += wgt * acc[v] * acc[v];
    }

    #pragma unroll
    for (int v = 0; v < 8; ++v) redE[w][v + 8 * lh][ln] = eacc[v];
    __syncthreads();
    if (tid < 16) {
        float s = 0.f;
        for (int ww = 0; ww < 4; ++ww)
            for (int nn = 0; nn < 16; ++nn)
                s += redE[ww][tid][nn];
        epc[((size_t)(b * NH + hb + tid)) * NJ + j] = s;
    }
}

// ---------------------------------------------------------------- kernel 4
// px pooled energies (same weighted-sum form as epc; scale factors cancel).
__global__ void k_px(const float* __restrict__ Xn, float* __restrict__ ex2) {
    __shared__ float red[256];
    int j = blockIdx.x, b = blockIdx.y, tid = threadIdx.x;
    const float* x = Xn + (size_t)b * NN;
    int tj = 2000 * j;
    float s = 0.f;
    for (int tp = tid; tp < TJLEN; tp += 256) {
        float v = x[tj + tp];           // tj+tp <= 64261 < 65536
        s += e2weight(tp) * v * v;
    }
    red[tid] = s; __syncthreads();
    for (int st = 128; st > 0; st >>= 1) {
        if (tid < st) red[tid] += red[tid + st];
        __syncthreads();
    }
    if (tid == 0) ex2[b * NJ + j] = red[0];
}

// ---------------------------------------------------------------- kernel 5
// mask[b,h] = all_j( px[j] >= pc[j] )  <=>  all_j( ex2[j]*Sc >= epc[j]*Sx ),
// then gate the filters already sitting in d_out.
__global__ void k_mask(const float* __restrict__ ex2,
                       const float* __restrict__ epc,
                       float* __restrict__ out) {
    int bid = blockIdx.x;
    int b = bid >> 9;
    int tid = threadIdx.x;
    const float* a = ex2 + b * NJ;
    const float* c = epc + (size_t)bid * NJ;
    float Sx = 0.f, Sc = 0.f;
    for (int jj = 0; jj < NJ; ++jj) { Sx += a[jj]; Sc += c[jj]; }
    bool ok = true;
    for (int jj = 0; jj < NJ; ++jj) ok = ok && (a[jj] * Sc >= c[jj] * Sx);
    float mf = ok ? 1.f : 0.f;
    float* o = out + (size_t)bid * KZF;
    for (int l = tid; l < KZF; l += 256) o[l] *= mf;
}

// ----------------------------------------------------------------
extern "C" void kernel_launch(void* const* d_in, const int* in_sizes, int n_in,
                              void* d_out, int out_size, void* d_ws, size_t ws_size,
                              hipStream_t stream) {
    const float* X = (const float*)d_in[0];
    float* out = (float*)d_out;
    char* ws = (char*)d_ws;

    size_t off = 0;
    float* Xn = (float*)(ws + off);          off += (size_t)NB * NN * sizeof(float);      // 2 MB
    h16* Ff16 = (h16*)(ws + off);            off += (size_t)NB * NH * 1024 * sizeof(h16); // 8 MB
    float* epc = (float*)(ws + off);         off += (size_t)NB * NH * NJ * sizeof(float); // 512 KB
    float* ex2 = (float*)(ws + off);         off += (size_t)NB * NJ * sizeof(float);      // 1 KB
    (void)ws_size; (void)in_sizes; (void)n_in; (void)out_size;

    k_norm<<<NB, 256, 0, stream>>>(X, Xn);
    k_autocov<<<NB * NH, 256, 0, stream>>>(Xn, out, Ff16);
    dim3 g3(NJ, NH / 16, NB);
    k_convpool<<<g3, 128, 0, stream>>>(Xn, Ff16, epc);
    dim3 gp(NJ, NB);
    k_px<<<gp, 256, 0, stream>>>(Xn, ex2);
    k_mask<<<NB * NH, 256, 0, stream>>>(ex2, epc, out);
}
